// DisentangledSelfAttention_69587060130238
// MI455X (gfx1250) — matrix operand order repaired and checked by reading, not compiled, not run on hardware
//
#include <hip/hip_runtime.h>
#include <hip/hip_bf16.h>
#include <math.h>

// ---------------------------------------------------------------------------
// DeBERTa disentangled self-attention for MI455X (gfx1250, wave32, WMMA)
// B=1, N=2048, H=1024, NH=16, dh=64, span=512, 2S=1024
// ---------------------------------------------------------------------------

typedef __attribute__((ext_vector_type(16))) __bf16 v16bf;
typedef __attribute__((ext_vector_type(8)))  __bf16 v8bf;
typedef __attribute__((ext_vector_type(8)))  float  v8f;

#define NN   2048
#define HH   1024
#define NHD  16
#define DH   64
#define SS   1024   /* 2*span */

__device__ __forceinline__ v8f wmma_bf16(v16bf a, v16bf b, v8f c) {
  return __builtin_amdgcn_wmma_f32_16x16x32_bf16(false, a, false, b, (short)0, c,
                                                 false, false);
}

// A-fragment (16x32 bf16): lane&15 = row; lanes<16 carry K {0..15},
// lanes>=16 carry K {16..31}. Caller pre-offsets pointer by (lane&16?16:0).
__device__ __forceinline__ v16bf afrag(const __bf16* p) {
  v8bf lo = *(const v8bf*)(p);
  v8bf hi = *(const v8bf*)(p + 8);
  return __builtin_shufflevector(lo, hi, 0, 1, 2, 3, 4, 5, 6, 7,
                                         8, 9, 10, 11, 12, 13, 14, 15);
}

// half-wave (16-lane) reductions for wave32
__device__ __forceinline__ float half_max(float v) {
  v = fmaxf(v, __shfl_xor(v, 1, 32));
  v = fmaxf(v, __shfl_xor(v, 2, 32));
  v = fmaxf(v, __shfl_xor(v, 4, 32));
  v = fmaxf(v, __shfl_xor(v, 8, 32));
  return v;
}
__device__ __forceinline__ float half_sum(float v) {
  v += __shfl_xor(v, 1, 32);
  v += __shfl_xor(v, 2, 32);
  v += __shfl_xor(v, 4, 32);
  v += __shfl_xor(v, 8, 32);
  return v;
}

// ---------------------------------------------------------------------------
// fp32 -> bf16 conversion (4 elems / thread)
// ---------------------------------------------------------------------------
__global__ __launch_bounds__(256) void cvt_bf16_kernel(const float* __restrict__ s,
                                                       __bf16* __restrict__ d, int n4) {
  int i = (blockIdx.x * 256 + threadIdx.x) * 4;
  if (i < n4) {
    float4 v = *(const float4*)(s + i);
    d[i + 0] = (__bf16)v.x;
    d[i + 1] = (__bf16)v.y;
    d[i + 2] = (__bf16)v.z;
    d[i + 3] = (__bf16)v.w;
  }
}

// ---------------------------------------------------------------------------
// core 16(M) x 64(O) wave tile, compile-time contraction KDIM.
// A[M][lda], B[O][ldb] row-major (B rows are GEMM columns: C = A * B^T).
// Fully unrolled: compile-time trip count lets the scheduler rename registers
// across iterations, hoisting iteration i+1's fragment loads above iteration
// i's WMMA chain (software pipelining with no register copies, no branches).
// ---------------------------------------------------------------------------
template <int KDIM>
__device__ __forceinline__ void mma_16x64(const __bf16* __restrict__ A,
                                          const __bf16* __restrict__ B,
                                          int lda, int ldb, int m0, int o0,
                                          int lane, v8f acc[4]) {
  const __bf16* arow = A + (size_t)(m0 + (lane & 15)) * lda + ((lane & 16) ? 16 : 0);
  const __bf16* brow = B + (size_t)(o0 + (lane & 15)) * ldb + ((lane & 16) ? 16 : 0);
#pragma unroll
  for (int k0 = 0; k0 < KDIM; k0 += 32) {
    v16bf a = afrag(arow + k0);
    v16bf b0 = *(const v16bf*)(brow + k0);
    v16bf b1 = *(const v16bf*)(brow + (size_t)16 * ldb + k0);
    v16bf b2 = *(const v16bf*)(brow + (size_t)32 * ldb + k0);
    v16bf b3 = *(const v16bf*)(brow + (size_t)48 * ldb + k0);
    acc[0] = wmma_bf16(a, b0, acc[0]);
    acc[1] = wmma_bf16(a, b1, acc[1]);
    acc[2] = wmma_bf16(a, b2, acc[2]);
    acc[3] = wmma_bf16(a, b3, acc[3]);
  }
}

// ---------------------------------------------------------------------------
// QKV projection: qp = hs @ Win^T, split per head (q|k|v blocks of 64 within
// each 192-channel head slice), q scaled, V stored transposed [h][d][n].
// grid(3072/64, 2048/(16*8)), block 256
// ---------------------------------------------------------------------------
__global__ __launch_bounds__(256)
void gemm_qkv_kernel(const __bf16* __restrict__ A, const __bf16* __restrict__ W,
                     const float* __restrict__ qb, const float* __restrict__ vb,
                     __bf16* __restrict__ Q, __bf16* __restrict__ K,
                     __bf16* __restrict__ Vt, float inv_scale) {
  int lane = threadIdx.x & 31, wv = threadIdx.x >> 5;
  int m0 = (blockIdx.y * 8 + wv) * 16;
  int o0 = blockIdx.x * 64;
  v8f acc[4] = {v8f{}, v8f{}, v8f{}, v8f{}};
  mma_16x64<HH>(A, W, HH, HH, m0, o0, lane, acc);

  int n = lane & 15, hf = (lane >> 4) & 1;
#pragma unroll
  for (int t = 0; t < 4; ++t) {
#pragma unroll
    for (int j = 0; j < 8; ++j) {
      int o = o0 + t * 16 + n;
      int m = m0 + j + hf * 8;
      int h = o / 192, r = o % 192;
      int d = r & 63, seg = r >> 6;
      float v = acc[t][j];
      if (seg == 0)
        Q[((size_t)h * NN + m) * DH + d] = (__bf16)((v + qb[h * DH + d]) * inv_scale);
      else if (seg == 1)
        K[((size_t)h * NN + m) * DH + d] = (__bf16)v;
      else
        Vt[((size_t)h * DH + d) * NN + m] = (__bf16)(v + vb[h * DH + d]);
    }
  }
}

// ---------------------------------------------------------------------------
// pos projection: out[h][s][d] = (rel @ Wp^T + bias) * mul   grid(16, 8)
// ---------------------------------------------------------------------------
__global__ __launch_bounds__(256)
void gemm_pos_kernel(const __bf16* __restrict__ A, const __bf16* __restrict__ W,
                     const float* __restrict__ bias, __bf16* __restrict__ P,
                     float mul) {
  int lane = threadIdx.x & 31, wv = threadIdx.x >> 5;
  int m0 = (blockIdx.y * 8 + wv) * 16;
  int o0 = blockIdx.x * 64;
  v8f acc[4] = {v8f{}, v8f{}, v8f{}, v8f{}};
  mma_16x64<HH>(A, W, HH, HH, m0, o0, lane, acc);

  int n = lane & 15, hf = (lane >> 4) & 1;
#pragma unroll
  for (int t = 0; t < 4; ++t) {
#pragma unroll
    for (int j = 0; j < 8; ++j) {
      int o = o0 + t * 16 + n;
      int m = m0 + j + hf * 8;
      float v = acc[t][j];
      if (bias) v += bias[o];
      P[((size_t)(o >> 6) * SS + m) * DH + (o & 63)] = (__bf16)(v * mul);
    }
  }
}

// ---------------------------------------------------------------------------
// batched head GEMM (K=64): Out[h][m][s] = A[h][m][:] . B[h][s][:]
// A:[NH][2048][64]  B:[NH][1024][64]  Out bf16 [NH][2048][1024]
// grid(1024/64, 2048/(16*8), NH)
// ---------------------------------------------------------------------------
__global__ __launch_bounds__(256)
void gemm_rel_kernel(const __bf16* __restrict__ Ah, const __bf16* __restrict__ Bh,
                     __bf16* __restrict__ Out) {
  int lane = threadIdx.x & 31, wv = threadIdx.x >> 5;
  int h = blockIdx.z;
  int m0 = (blockIdx.y * 8 + wv) * 16;
  int s0 = blockIdx.x * 64;
  const __bf16* A = Ah + (size_t)h * NN * DH;
  const __bf16* B = Bh + (size_t)h * SS * DH;
  v8f acc[4] = {v8f{}, v8f{}, v8f{}, v8f{}};
  mma_16x64<DH>(A, B, DH, DH, m0, s0, lane, acc);

  int n = lane & 15, hf = (lane >> 4) & 1;
  __bf16* out = Out + (size_t)h * NN * SS;
#pragma unroll
  for (int t = 0; t < 4; ++t) {
#pragma unroll
    for (int j = 0; j < 8; ++j)
      out[(size_t)(m0 + j + hf * 8) * SS + s0 + t * 16 + n] = (__bf16)acc[t][j];
  }
}

// ---------------------------------------------------------------------------
// fused flash attention: scores(WMMA) + gathered c2p/p2c bias + mask +
// online softmax + P@V (WMMA, P re-laid-out C->A via per-wave LDS tile).
// All loads independent of the score WMMAs (gathers, masks, V fragments) are
// issued *before* the WMMA chain so their latency hides under matrix math.
// grid(2048/(16*8), NH), block 256 (8 waves; wave owns a 16-row q tile)
// ---------------------------------------------------------------------------
__global__ __launch_bounds__(256)
void flash_attn_kernel(const __bf16* __restrict__ Q, const __bf16* __restrict__ K,
                       const __bf16* __restrict__ Vt,
                       const __bf16* __restrict__ C2P, const __bf16* __restrict__ P2C,
                       const int* __restrict__ mask, float* __restrict__ out) {
  int lane = threadIdx.x & 31, wv = threadIdx.x >> 5;
  int h = blockIdx.y;
  int q0 = (blockIdx.x * 8 + wv) * 16;
  int n = lane & 15, hf = (lane >> 4) & 1;

  __shared__ __bf16 plds[8][16][32];

  const __bf16* Qh = Q + (size_t)h * NN * DH;
  const __bf16* Kh = K + (size_t)h * NN * DH;
  const __bf16* Vh = Vt + (size_t)h * DH * NN;
  const __bf16* Ch = C2P + (size_t)h * NN * SS;
  const __bf16* Ph = P2C + (size_t)h * NN * SS;

  // Q A-fragments for d=0..31 and d=32..63 (held for entire K loop)
  const __bf16* qrow = Qh + (size_t)(q0 + n) * DH + ((lane & 16) ? 16 : 0);
  v16bf qa0 = afrag(qrow);
  v16bf qa1 = afrag(qrow + 32);

  v8f o[4] = {v8f{}, v8f{}, v8f{}, v8f{}};
  float mi[8], li[8];
#pragma unroll
  for (int j = 0; j < 8; ++j) { mi[j] = -__builtin_inff(); li[j] = 0.f; }

  for (int k0 = 0; k0 < NN; k0 += 32) {
    // ---- issue ALL independent loads first -----------------------------
    const __bf16* kr = Kh + (size_t)(k0 + n) * DH + ((lane & 16) ? 16 : 0);
    v16bf kb0 = *(const v16bf*)(kr);
    v16bf kb1 = *(const v16bf*)(kr + 32);
    v16bf kb2 = *(const v16bf*)(kr + 16 * DH);
    v16bf kb3 = *(const v16bf*)(kr + 16 * DH + 32);

    v16bf vbf[4];
#pragma unroll
    for (int t = 0; t < 4; ++t)
      vbf[t] = *(const v16bf*)(Vh + (size_t)(t * 16 + n) * NN + k0 +
                               ((lane & 16) ? 16 : 0));

    float g0[8], g1[8];
    int mk0[8], mk1[8];
#pragma unroll
    for (int j = 0; j < 8; ++j) {
      int qq = q0 + j + hf * 8;
      int kA = k0 + n, kB = k0 + 16 + n;
      int i0 = min(max(qq - kA + 512, 0), SS - 1);
      int i1 = min(max(qq - kB + 512, 0), SS - 1);
      g0[j] = (float)Ch[(size_t)qq * SS + i0] + (float)Ph[(size_t)kA * SS + i0];
      g1[j] = (float)Ch[(size_t)qq * SS + i1] + (float)Ph[(size_t)kB * SS + i1];
      mk0[j] = mask[(size_t)qq * NN + kA];
      mk1[j] = mask[(size_t)qq * NN + kB];
    }

    // ---- S = Q K^T for two 16-key tiles (overlaps gather latency) ------
    v8f s0a = {}, s1a = {};
    s0a = wmma_bf16(qa0, kb0, s0a);
    s0a = wmma_bf16(qa1, kb1, s0a);
    s1a = wmma_bf16(qa0, kb2, s1a);
    s1a = wmma_bf16(qa1, kb3, s1a);

    // ---- disentangled bias, mask, online softmax stats -----------------
    float p0[8], p1[8], alpha[8];
#pragma unroll
    for (int j = 0; j < 8; ++j) {
      float s0 = s0a[j] + g0[j];
      float s1 = s1a[j] + g1[j];
      if (!mk0[j]) s0 = -__builtin_inff();
      if (!mk1[j]) s1 = -__builtin_inff();
      float mx = half_max(fmaxf(s0, s1));
      float nm = fmaxf(mi[j], mx);
      alpha[j] = (mi[j] == -__builtin_inff()) ? 0.f : __expf(mi[j] - nm);
      p0[j] = __expf(s0 - nm);
      p1[j] = __expf(s1 - nm);
      li[j] = li[j] * alpha[j] + half_sum(p0[j] + p1[j]);
      mi[j] = nm;
    }

    // ---- rescale running output ----------------------------------------
#pragma unroll
    for (int t = 0; t < 4; ++t)
#pragma unroll
      for (int j = 0; j < 8; ++j) o[t][j] *= alpha[j];

    // ---- C-layout P -> A-layout via per-wave LDS tile ------------------
#pragma unroll
    for (int j = 0; j < 8; ++j) {
      int m = j + hf * 8;
      plds[wv][m][n] = (__bf16)p0[j];
      plds[wv][m][16 + n] = (__bf16)p1[j];
    }
    asm volatile("s_wait_dscnt 0" ::: "memory");  // LDS RAW fence (CDNA5 split cnt)
    v16bf pa = afrag(&plds[wv][n][(lane & 16) ? 16 : 0]);

    // ---- O += P V  (V fragments already resident) ----------------------
#pragma unroll
    for (int t = 0; t < 4; ++t) o[t] = wmma_bf16(pa, vbf[t], o[t]);
  }

  // ---- normalize and write ctx[q][h*64+d] -------------------------------
  float inv[8];
#pragma unroll
  for (int j = 0; j < 8; ++j) inv[j] = 1.f / li[j];
#pragma unroll
  for (int t = 0; t < 4; ++t)
#pragma unroll
    for (int j = 0; j < 8; ++j)
      out[(size_t)(q0 + j + hf * 8) * HH + h * DH + t * 16 + n] = o[t][j] * inv[j];
}

// ---------------------------------------------------------------------------
// host side
// ---------------------------------------------------------------------------
extern "C" void kernel_launch(void* const* d_in, const int* in_sizes, int n_in,
                              void* d_out, int out_size, void* d_ws, size_t ws_size,
                              hipStream_t stream) {
  const float* hs        = (const float*)d_in[0];
  const int*   amask     = (const int*)d_in[1];
  /* d_in[2] relative_pos: index computed analytically (rpos = q-k) */
  const float* rel_emb   = (const float*)d_in[3];  // [1024,1024] == full slice
  const float* in_proj_w = (const float*)d_in[4];  // [3072,1024]
  const float* q_bias    = (const float*)d_in[5];
  const float* v_bias    = (const float*)d_in[6];
  const float* pos_w     = (const float*)d_in[7];
  const float* pos_q_w   = (const float*)d_in[8];
  const float* pos_q_b   = (const float*)d_in[9];
  float* out = (float*)d_out;

  char* ws = (char*)d_ws;
  size_t off = 0;
  auto alloc = [&](size_t bytes) {
    void* p = ws + off;
    off += (bytes + 255) & ~(size_t)255;
    return p;
  };
  __bf16* hs_bf  = (__bf16*)alloc((size_t)NN * HH * 2);
  __bf16* win_bf = (__bf16*)alloc((size_t)3 * HH * HH * 2);
  __bf16* rel_bf = (__bf16*)alloc((size_t)SS * HH * 2);
  __bf16* wpk_bf = (__bf16*)alloc((size_t)HH * HH * 2);
  __bf16* wpq_bf = (__bf16*)alloc((size_t)HH * HH * 2);
  __bf16* Qb     = (__bf16*)alloc((size_t)NHD * NN * DH * 2);
  __bf16* Kb     = (__bf16*)alloc((size_t)NHD * NN * DH * 2);
  __bf16* Vtb    = (__bf16*)alloc((size_t)NHD * DH * NN * 2);
  __bf16* posK   = (__bf16*)alloc((size_t)NHD * SS * DH * 2);
  __bf16* posQ   = (__bf16*)alloc((size_t)NHD * SS * DH * 2);
  __bf16* c2p    = (__bf16*)alloc((size_t)NHD * NN * SS * 2);
  __bf16* p2c    = (__bf16*)alloc((size_t)NHD * NN * SS * 2);

  const float inv_scale = 1.0f / sqrtf((float)(DH * 3));

  auto cvt = [&](const float* s, __bf16* d, int n) {
    cvt_bf16_kernel<<<(n / 4 + 255) / 256, 256, 0, stream>>>(s, d, n);
  };
  cvt(hs, hs_bf, NN * HH);
  cvt(in_proj_w, win_bf, 3 * HH * HH);
  cvt(rel_emb, rel_bf, SS * HH);
  cvt(pos_w, wpk_bf, HH * HH);
  cvt(pos_q_w, wpq_bf, HH * HH);

  gemm_qkv_kernel<<<dim3(3 * HH / 64, NN / 128), 256, 0, stream>>>(
      hs_bf, win_bf, q_bias, v_bias, Qb, Kb, Vtb, inv_scale);

  gemm_pos_kernel<<<dim3(HH / 64, SS / 128), 256, 0, stream>>>(
      rel_bf, wpk_bf, nullptr, posK, 1.0f);
  gemm_pos_kernel<<<dim3(HH / 64, SS / 128), 256, 0, stream>>>(
      rel_bf, wpq_bf, pos_q_b, posQ, inv_scale);

  gemm_rel_kernel<<<dim3(SS / 64, NN / 128, NHD), 256, 0, stream>>>(Qb, posK, c2p);
  gemm_rel_kernel<<<dim3(SS / 64, NN / 128, NHD), 256, 0, stream>>>(Kb, posQ, p2c);

  flash_attn_kernel<<<dim3(NN / 128, NHD), 256, 0, stream>>>(
      Qb, Kb, Vtb, c2p, p2c, amask, out);
}
